// GPTAttention_89816356094752
// MI455X (gfx1250) — matrix operand order repaired and checked by reading, not compiled, not run on hardware
//
#include <hip/hip_runtime.h>
#include <hip/hip_bf16.h>

// ---------------------------------------------------------------------------
// GPT causal self-attention block for MI455X (gfx1250, wave32, WMMA).
// All matmuls run on v_wmma_f32_16x16x32_bf16 (bf16 inputs, fp32 accum).
// Pipeline (separate kernels, serialized on `stream`):
//   1) fp32 -> bf16 conversion of x, w_attn, w_proj
//   2) QKV GEMM (bias fused, register-double-buffered LDS tiles),
//      scatter Q,K as [BH][T][D], V transposed [BH][D][T]
//   3) flash-attention: block-cooperative K/V staging via gfx1250
//      global_load_async_to_lds_b128 (ASYNCcnt-tracked), online softmax,
//      y -> bf16 [B*T][C]
//   4) output-proj GEMM (bias fused) -> fp32 d_out
// ---------------------------------------------------------------------------

typedef __attribute__((ext_vector_type(16))) __bf16 v16bf;
typedef __attribute__((ext_vector_type(8)))  float  v8f;

typedef unsigned short u16;
typedef unsigned int   u32;

union Frag16 { v16bf v; u32 u[8]; };

#define WMMA_BF16(Af, Bf, Cv) \
    __builtin_amdgcn_wmma_f32_16x16x32_bf16(false, (Af), false, (Bf), (short)0, (Cv), false, false)

// gfx1250 async global->LDS copy (ASYNCcnt-tracked).
// Probe-confirmed signature: (int4-vec addrspace(1)*, int4-vec addrspace(3)*,
// imm offset, imm cpol). Guarded fallback: plain vector copy.
#if defined(__has_builtin)
#if __has_builtin(__builtin_amdgcn_global_load_async_to_lds_b128)
#define HAVE_ASYNC_LDS 1
#endif
#endif

typedef int v4i_vs __attribute__((vector_size(16)));
typedef __attribute__((address_space(1))) v4i_vs as1_v4i;
typedef __attribute__((address_space(3))) v4i_vs as3_v4i;

#ifdef HAVE_ASYNC_LDS
#define ASYNC_CP16(dst, src)                                                  \
    __builtin_amdgcn_global_load_async_to_lds_b128(                           \
        (as1_v4i*)(src), (as3_v4i*)(dst), 0, 0)
#if __has_builtin(__builtin_amdgcn_s_wait_asynccnt)
#define ASYNC_WAIT() __builtin_amdgcn_s_wait_asynccnt(0)
#else
#define ASYNC_WAIT() asm volatile("s_wait_asynccnt 0" ::: "memory")
#endif
#else
#define ASYNC_CP16(dst, src) (*(uint4*)(dst) = *(const uint4*)(src))
#define ASYNC_WAIT()
#endif

__device__ __forceinline__ u16 f2bf(float f) {
    u32 u = __float_as_uint(f);
    u32 r = (u + 0x7FFFu + ((u >> 16) & 1u)) >> 16;  // round-to-nearest-even
    return (u16)r;
}

#define LDT 40  // padded LDS row stride (bf16 elements): 32 data + 8 pad

// ---------------------------------------------------------------------------
// 1) elementwise fp32 -> bf16
// ---------------------------------------------------------------------------
__global__ void cvt_f32_bf16(const float* __restrict__ in, u16* __restrict__ out, int n) {
    int i = blockIdx.x * blockDim.x + threadIdx.x;
    int stride = gridDim.x * blockDim.x;
    for (; i < n; i += stride) out[i] = f2bf(in[i]);
}

// ---------------------------------------------------------------------------
// 2)/4) tiled bf16 GEMM:  C[M,N] = A[M,K] * B[K,N] + bias[N]
//   block tile 128x128, BK=32, 256 threads = 8 waves in a 2(M) x 4(N) grid,
//   each wave: 64x32 -> 4x2 accumulators, 8 v_wmma per K-step.
//   Register double-buffering: next tile is prefetched into VGPRs while the
//   matrix pipe consumes the current LDS tile.
// mode 0: QKV epilogue (scatter bf16 Q/K/Vt);  mode 1: fp32 store to outf.
// ---------------------------------------------------------------------------
__global__ __launch_bounds__(256) void gemm_bf16(
    const u16* __restrict__ A, const u16* __restrict__ B, const float* __restrict__ bias,
    int K, int N, int mode,
    u16* __restrict__ qb, u16* __restrict__ kb, u16* __restrict__ vtb,
    float* __restrict__ outf)
{
    __shared__ __align__(16) u16 aS[128 * LDT];   // A tile, row-major [m][k]
    __shared__ __align__(16) u16 bS[128 * LDT];   // B tile, transposed [n][k]

    const int tid  = threadIdx.x;
    const int lane = tid & 31;
    const int wid  = tid >> 5;
    const int wm   = wid & 1;     // wave position along M (2)
    const int wn   = wid >> 1;    // wave position along N (4)
    const int nl   = lane & 15;
    const int hh   = lane >> 4;

    const int m0 = blockIdx.y * 128;
    const int n0 = blockIdx.x * 128;

    // cooperative tile-load mapping
    const int ar  = tid >> 2;         // A row 0..63 (+64)
    const int ac  = (tid & 3) * 8;    // A col 0,8,16,24
    const int bkr = tid >> 4;         // B k-row 0..15 (+16)
    const int bc  = (tid & 15) * 8;   // B col 0..120

    const u16* aSrc = A + (size_t)(m0 + ar) * K + ac;        // + kt
    const u16* bSrc = B + (size_t)bkr * N + n0 + bc;         // + kt*N

    v8f acc[4][2];
    #pragma unroll
    for (int mi = 0; mi < 4; ++mi)
        #pragma unroll
        for (int ni = 0; ni < 2; ++ni)
            #pragma unroll
            for (int r = 0; r < 8; ++r) acc[mi][ni][r] = 0.0f;

    // prefetch tile 0 into registers
    uint4 ra0 = *(const uint4*)(aSrc);
    uint4 ra1 = *(const uint4*)(aSrc + (size_t)64 * K);
    uint4 rb0 = *(const uint4*)(bSrc);
    uint4 rb1 = *(const uint4*)(bSrc + (size_t)16 * N);

    for (int kt = 0; kt < K; kt += 32) {
        // commit current tile registers to LDS
        *(uint4*)&aS[ar * LDT + ac]        = ra0;
        *(uint4*)&aS[(ar + 64) * LDT + ac] = ra1;
        {
            const u16* p0 = (const u16*)&rb0;
            const u16* p1 = (const u16*)&rb1;
            #pragma unroll
            for (int j = 0; j < 8; ++j) {           // transpose into [n][k]
                bS[(bc + j) * LDT + bkr]      = p0[j];
                bS[(bc + j) * LDT + bkr + 16] = p1[j];
            }
        }
        __syncthreads();

        // issue next-tile global loads early (overlap with WMMA below)
        if (kt + 32 < K) {
            const u16* aN = aSrc + kt + 32;
            const u16* bN = bSrc + (size_t)(kt + 32) * N;
            ra0 = *(const uint4*)(aN);
            ra1 = *(const uint4*)(aN + (size_t)64 * K);
            rb0 = *(const uint4*)(bN);
            rb1 = *(const uint4*)(bN + (size_t)16 * N);
        }

        // A fragment: lane m = nl, element e -> K = e + 16h
        Frag16 af[4], bf[2];
        #pragma unroll
        for (int mi = 0; mi < 4; ++mi) {
            const int mrow = wm * 64 + mi * 16 + nl;
            #pragma unroll
            for (int p = 0; p < 8; ++p) {
                const int kp = 2 * p + 16 * hh;
                af[mi].u[p] = *(const u32*)&aS[mrow * LDT + kp];
            }
        }
        // B fragment: lane n = nl, element e -> K = e + 16h
        #pragma unroll
        for (int ni = 0; ni < 2; ++ni) {
            const int nrow = wn * 32 + ni * 16 + nl;
            #pragma unroll
            for (int p = 0; p < 8; ++p) {
                const int kp = 2 * p + 16 * hh;
                bf[ni].u[p] = *(const u32*)&bS[nrow * LDT + kp];
            }
        }
        #pragma unroll
        for (int mi = 0; mi < 4; ++mi)
            #pragma unroll
            for (int ni = 0; ni < 2; ++ni)
                acc[mi][ni] = WMMA_BF16(af[mi].v, bf[ni].v, acc[mi][ni]);

        __syncthreads();
    }

    // epilogue; C layout: row M = r + 8h, col N = nl
    #pragma unroll
    for (int mi = 0; mi < 4; ++mi) {
        #pragma unroll
        for (int ni = 0; ni < 2; ++ni) {
            #pragma unroll
            for (int r = 0; r < 8; ++r) {
                const int mg = m0 + wm * 64 + mi * 16 + r + 8 * hh;
                const int ng = n0 + wn * 32 + ni * 16 + nl;
                const float v = acc[mi][ni][r] + bias[ng];
                if (mode == 0) {
                    const int which = (ng >= 5120) ? 2 : ((ng >= 2560) ? 1 : 0);
                    const int c    = ng - which * 2560;
                    const int head = c >> 7, d = c & 127;
                    const int b    = mg >> 11, t = mg & 2047;
                    const size_t bh = (size_t)(b * 20 + head);
                    if (which == 0)      qb [(bh * 2048 + t) * 128 + d] = f2bf(v);
                    else if (which == 1) kb [(bh * 2048 + t) * 128 + d] = f2bf(v);
                    else                 vtb[(bh * 128 + d) * 2048 + t] = f2bf(v); // transposed
                } else {
                    outf[(size_t)mg * N + ng] = v;
                }
            }
        }
    }
}

// ---------------------------------------------------------------------------
// 3) flash attention, block-cooperative K/V staging.
//    Block = 8 waves x 16 query rows = 128 rows. All waves iterate over the
//    block's causal key range; K block (32x128) and Vt block (128x32) are
//    staged into LDS once per iteration via async-to-LDS, then each wave runs
//    S = Q*K^T (8 wmma), online softmax (shfl_xor row reductions in 16-lane
//    halves), P relayout via per-wave LDS tile, and O += P*V (8 wmma).
//    Waves past their causal range skip compute but keep barriers uniform.
// ---------------------------------------------------------------------------
#define KLD 136  // LDS row stride for K tile [t][d]: 128 data + 8 pad

__global__ __launch_bounds__(256) void attn_fa(
    const u16* __restrict__ qb, const u16* __restrict__ kb, const u16* __restrict__ vtb,
    u16* __restrict__ yb)
{
    __shared__ __align__(16) u16 kS[32 * KLD];      // K block  [t_local][d]
    __shared__ __align__(16) u16 vS[128 * LDT];     // Vt block [d][t_local]
    __shared__ __align__(16) u16 pS[8 * 16 * LDT];  // per-wave 16x32 P tile

    const int bh   = blockIdx.x;           // 0..39
    const int b    = bh / 20;
    const int head = bh % 20;
    const int tid  = threadIdx.x;
    const int lane = tid & 31;
    const int wid  = tid >> 5;
    const int nl   = lane & 15;
    const int hh   = lane >> 4;

    const int qbase = blockIdx.y * 128 + wid * 16;

    const u16* qptr = qb  + (size_t)bh * 2048 * 128;
    const u16* kptr = kb  + (size_t)bh * 2048 * 128;
    const u16* vptr = vtb + (size_t)bh * 128  * 2048;

    // Q fragments (A-layout), d = 0..127 in 4 frags of K=32
    Frag16 aq[4];
    #pragma unroll
    for (int f = 0; f < 4; ++f) {
        const u16* qrow = qptr + (size_t)(qbase + nl) * 128 + f * 32;
        #pragma unroll
        for (int p = 0; p < 8; ++p) {
            const int kp = 2 * p + 16 * hh;
            aq[f].u[p] = *(const u32*)(qrow + kp);
        }
    }

    v8f o[8];
    float mrow[8], lrow[8];
    #pragma unroll
    for (int c = 0; c < 8; ++c)
        #pragma unroll
        for (int r = 0; r < 8; ++r) o[c][r] = 0.0f;
    #pragma unroll
    for (int r = 0; r < 8; ++r) { mrow[r] = -1e30f; lrow[r] = 0.0f; }

    const float scale = 0.08838834764831845f;  // 1/sqrt(128)
    const int nkb_blk = blockIdx.y * 4 + 4;    // key blocks for whole block
    const int nkb_my  = (qbase + 16 + 31) >> 5;// key blocks this wave needs
    u16* myP = &pS[wid * 16 * LDT];

    // cooperative staging mapping
    const int skr = tid >> 4;         // K rows 0..15 (+16)
    const int skc = (tid & 15) * 8;   // K cols 0..120
    const int svr = tid >> 2;         // V rows 0..63 (+64)
    const int svc = (tid & 3) * 8;    // V cols 0,8,16,24

    for (int kbi = 0; kbi < nkb_blk; ++kbi) {
        const int kbase = kbi * 32;

        // ---- stage K (32 x 128) and Vt (128 x 32) into LDS (async) ----
        ASYNC_CP16(&kS[skr * KLD + skc],        kptr + (size_t)(kbase + skr) * 128 + skc);
        ASYNC_CP16(&kS[(skr + 16) * KLD + skc], kptr + (size_t)(kbase + skr + 16) * 128 + skc);
        ASYNC_CP16(&vS[svr * LDT + svc],        vptr + (size_t)svr * 2048 + kbase + svc);
        ASYNC_CP16(&vS[(svr + 64) * LDT + svc], vptr + (size_t)(svr + 64) * 2048 + kbase + svc);
        ASYNC_WAIT();
        __syncthreads();

        if (kbi < nkb_my) {
            // ---- S = Q * K^T (two 16-col tiles) ----
            v8f s0, s1;
            #pragma unroll
            for (int r = 0; r < 8; ++r) { s0[r] = 0.0f; s1[r] = 0.0f; }
            #pragma unroll
            for (int f = 0; f < 4; ++f) {
                Frag16 bk0, bk1;
                #pragma unroll
                for (int p = 0; p < 8; ++p) {
                    const int kp = f * 32 + 2 * p + 16 * hh;   // d index
                    bk0.u[p] = *(const u32*)&kS[nl * KLD + kp];
                    bk1.u[p] = *(const u32*)&kS[(nl + 16) * KLD + kp];
                }
                s0 = WMMA_BF16(aq[f].v, bk0.v, s0);
                s1 = WMMA_BF16(aq[f].v, bk1.v, s1);
            }

            // ---- causal mask + online softmax (rows r+8h per lane) ----
            #pragma unroll
            for (int r = 0; r < 8; ++r) {
                const int qrow = qbase + r + 8 * hh;
                float a0 = s0[r] * scale;
                float a1 = s1[r] * scale;
                if (kbase + nl > qrow)      a0 = -1e30f;
                if (kbase + 16 + nl > qrow) a1 = -1e30f;
                float mx = fmaxf(a0, a1);
                #pragma unroll
                for (int off = 1; off < 16; off <<= 1)
                    mx = fmaxf(mx, __shfl_xor(mx, off, 32));
                const float mnew  = fmaxf(mrow[r], mx);
                const float alpha = __expf(mrow[r] - mnew);
                const float p0 = __expf(a0 - mnew);
                const float p1 = __expf(a1 - mnew);
                float rs = p0 + p1;
                #pragma unroll
                for (int off = 1; off < 16; off <<= 1)
                    rs += __shfl_xor(rs, off, 32);
                lrow[r] = lrow[r] * alpha + rs;
                mrow[r] = mnew;
                #pragma unroll
                for (int c = 0; c < 8; ++c) o[c][r] *= alpha;
                // park P in LDS (C-layout -> memory) for A-fragment reload
                myP[(r + 8 * hh) * LDT + nl]      = f2bf(p0);
                myP[(r + 8 * hh) * LDT + nl + 16] = f2bf(p1);
            }

            asm volatile("s_wait_dscnt 0" ::: "memory");  // LDS write->read, same wave

            // P as A-fragment (16x32)
            Frag16 pf;
            #pragma unroll
            for (int p = 0; p < 8; ++p) {
                const int kp = 2 * p + 16 * hh;
                pf.u[p] = *(const u32*)&myP[nl * LDT + kp];
            }

            // ---- O += P * V (from staged transposed V) ----
            #pragma unroll
            for (int c = 0; c < 8; ++c) {
                Frag16 vf;
                #pragma unroll
                for (int p = 0; p < 8; ++p) {
                    const int kp = 2 * p + 16 * hh;            // t_local index
                    vf.u[p] = *(const u32*)&vS[(c * 16 + nl) * LDT + kp];
                }
                o[c] = WMMA_BF16(pf.v, vf.v, o[c]);
            }
        }
        __syncthreads();
    }

    // normalize and emit y as bf16 [B*T][2560] (head-interleaved columns)
    #pragma unroll
    for (int r = 0; r < 8; ++r) {
        const float inv = 1.0f / lrow[r];
        const int t = qbase + r + 8 * hh;
        u16* yrow = yb + (size_t)(b * 2048 + t) * 2560 + head * 128;
        #pragma unroll
        for (int c = 0; c < 8; ++c)
            yrow[c * 16 + nl] = f2bf(o[c][r] * inv);
    }
}

// ---------------------------------------------------------------------------
// launch
// ---------------------------------------------------------------------------
extern "C" void kernel_launch(void* const* d_in, const int* in_sizes, int n_in,
                              void* d_out, int out_size, void* d_ws, size_t ws_size,
                              hipStream_t stream) {
    (void)in_sizes; (void)n_in; (void)out_size; (void)ws_size;
    const float* x      = (const float*)d_in[0];
    const float* w_attn = (const float*)d_in[1];
    const float* b_attn = (const float*)d_in[2];
    const float* w_proj = (const float*)d_in[3];
    const float* b_proj = (const float*)d_in[4];
    float* out = (float*)d_out;

    // workspace layout (bytes, 256-aligned)
    char* ws = (char*)d_ws;
    u16* xb  = (u16*)(ws + 0);           // x    bf16 [4096][2560]   20,971,520 B
    u16* wa  = (u16*)(ws + 20971520);    // Wqkv bf16 [2560][7680]   39,321,600 B
    u16* wp  = (u16*)(ws + 60293120);    // Wprj bf16 [2560][2560]   13,107,200 B
    u16* qb  = (u16*)(ws + 73400320);    // Q bf16 [40][2048][128]
    u16* kb  = (u16*)(ws + 94371840);    // K bf16 [40][2048][128]
    u16* vtb = (u16*)(ws + 115343360);   // V^T bf16 [40][128][2048]
    u16* yb  = (u16*)(ws + 136314880);   // y bf16 [4096][2560]  (end 157,286,400)

    cvt_f32_bf16<<<2048, 256, 0, stream>>>(x,      xb, 4096 * 2560);
    cvt_f32_bf16<<<2048, 256, 0, stream>>>(w_attn, wa, 2560 * 7680);
    cvt_f32_bf16<<<2048, 256, 0, stream>>>(w_proj, wp, 2560 * 2560);

    // QKV: [4096,2560] x [2560,7680]
    gemm_bf16<<<dim3(60, 32), 256, 0, stream>>>(xb, wa, b_attn, 2560, 7680, 0,
                                                qb, kb, vtb, nullptr);
    // attention: 40 (B*H) x 16 q-blocks of 128 rows
    attn_fa<<<dim3(40, 16), 256, 0, stream>>>(qb, kb, vtb, yb);
    // proj: [4096,2560] x [2560,2560] -> fp32 out
    gemm_bf16<<<dim3(20, 32), 256, 0, stream>>>(yb, wp, b_proj, 2560, 2560, 1,
                                                nullptr, nullptr, nullptr, out);
}
